// LinearAttentionFusedChunkKernel_28192165331363
// MI455X (gfx1250) — matrix operand order repaired and checked by reading, not compiled, not run on hardware
//
#include <hip/hip_runtime.h>
#include <hip/hip_bf16.h>

// ---------------- problem constants ----------------
#define B_   4
#define S_   4096
#define H_   16
#define D_   128
#define CHK  64
#define NT_  (S_ / CHK)         // 64 chunks
#define NSL  4                  // Dv slices per head
#define DV   (D_ / NSL)         // 32
#define SCALE 0.08838834764831843f   // 1/sqrt(128)

// padded LDS pitches (elements). All row pitches keep 16B alignment and
// shift banks between consecutive rows to avoid multi-way conflicts.
#define QP   136   // bf16 pitch for 64x128 Q/K tiles   (272B = 17*16)
#define KTP  72    // bf16 pitch for 128x64 K^T tile    (144B = 9*16)
#define VTP  72    // bf16 pitch for 32x64 V^T tile
#define SP_  72    // bf16 pitch for 64x64 S tile
#define HP   132   // f32 pitch for 32x128 state tile   (528B = 33*16)
#define HBP  136   // bf16 pitch for 32x128 state shadow

typedef __attribute__((ext_vector_type(16))) __bf16        v16bf;
typedef __attribute__((ext_vector_type(2)))  __bf16        v2bf;
typedef __attribute__((ext_vector_type(8)))  float         v8f;
typedef __attribute__((ext_vector_type(2)))  float         f32x2;
typedef __attribute__((ext_vector_type(4)))  unsigned int  u32x4;

union Frag16 {
    u32x4        q[2];
    unsigned int u[8];
    v16bf        v;
};

// f32 pair -> packed bf16. Prefer the HW packed convert; otherwise let the
// backend lower a 2-wide fptrunc (picks v_cvt_pk_bf16_f32 when available).
static __device__ __forceinline__ unsigned int pk2bf(float a, float b) {
#if __has_builtin(__builtin_amdgcn_cvt_pk_bf16_f32)
    v2bf r = __builtin_amdgcn_cvt_pk_bf16_f32(a, b);
    return __builtin_bit_cast(unsigned int, r);
#else
    f32x2 f = {a, b};
    v2bf  r = __builtin_convertvector(f, v2bf);
    return __builtin_bit_cast(unsigned int, r);
#endif
}

__global__ __launch_bounds__(256)
void linattn_fused_chunk_kernel(const float* __restrict__ q,
                                const float* __restrict__ k,
                                const float* __restrict__ v,
                                float* __restrict__ out) {
    // block -> (b, h, slice)
    const int slice = blockIdx.x % NSL;
    const int h     = (blockIdx.x / NSL) % H_;
    const int b     = blockIdx.x / (NSL * H_);

    const int tid    = threadIdx.x;
    const int wave   = tid >> 5;
    const int lane   = tid & 31;
    const int laneM  = lane & 15;       // N / M-low selector
    const int laneHi = lane >> 4;       // 0 or 1

    __shared__ __align__(16) unsigned short Qs[CHK * QP];     // Q chunk, bf16, scaled
    __shared__ __align__(16) unsigned short Ks[CHK * QP];     // K chunk, bf16, row-major
    __shared__ __align__(16) unsigned short Kt[D_ * KTP];     // K chunk transposed
    __shared__ __align__(16) unsigned short Vt[DV * VTP];     // V slice transposed
    __shared__ __align__(16) unsigned short Ss[CHK * SP_];    // masked scores, bf16
    __shared__ __align__(16) float          Ht[DV * HP];      // state slice, f32 master
    __shared__ __align__(16) unsigned short Hb[DV * HBP];     // state slice, bf16 shadow

    // zero state (both copies)
    for (int i = tid; i < DV * HP; i += 256) Ht[i] = 0.0f;
    for (int i = tid; i < DV * HBP; i += 256) Hb[i] = 0;

    const size_t rowStride = (size_t)H_ * D_;                 // 2048
    const size_t baseBH    = ((size_t)b * S_) * rowStride + (size_t)h * D_;

    for (int ct = 0; ct < NT_; ++ct) {
        const int s0 = ct * CHK;
        __syncthreads();   // previous chunk fully consumed before restaging

        // ---- stage Q, K (64x128) and V slice (64x32) as bf16, pairwise ----
        #pragma unroll 4
        for (int i = 0; i < 16; ++i) {
            int pidx = tid + 256 * i;         // 0..4095 float pairs
            int r = pidx >> 6;
            int c = (pidx & 63) << 1;
            size_t g = baseBH + (size_t)(s0 + r) * rowStride + c;
            f32x2 q2 = *(const f32x2*)&q[g];
            f32x2 k2 = *(const f32x2*)&k[g];
            *(unsigned int*)&Qs[r * QP + c] = pk2bf(q2.x * SCALE, q2.y * SCALE);
            unsigned int kp = pk2bf(k2.x, k2.y);
            *(unsigned int*)&Ks[r * QP + c] = kp;
            Kt[c * KTP + r]       = (unsigned short)(kp & 0xFFFFu);
            Kt[(c + 1) * KTP + r] = (unsigned short)(kp >> 16);
        }
        #pragma unroll
        for (int i = 0; i < 4; ++i) {
            int pidx = tid + 256 * i;         // 0..1023 float pairs
            int r = pidx >> 4;
            int c = (pidx & 15) << 1;
            size_t g = baseBH + (size_t)(s0 + r) * rowStride + slice * DV + c;
            f32x2 v2 = *(const f32x2*)&v[g];
            unsigned int vp = pk2bf(v2.x, v2.y);
            Vt[c * VTP + r]       = (unsigned short)(vp & 0xFFFFu);
            Vt[(c + 1) * VTP + r] = (unsigned short)(vp >> 16);
        }
        __syncthreads();

        // ---- phase 1: S = (Q*scale) @ K^T, causal mask, 16 tiles, 2/wave ----
        #pragma unroll
        for (int t = 0; t < 2; ++t) {
            const int tile = wave + t * 8;
            const int mt = tile >> 2, nt = tile & 3;
            v8f acc = 0.0f;
            #pragma unroll
            for (int kc = 0; kc < 4; ++kc) {
                const int arow = mt * 16 + laneM;
                const int akb  = kc * 32 + laneHi * 8;
                Frag16 fa;
                fa.q[0] = *(const u32x4*)&Qs[arow * QP + akb];
                fa.q[1] = *(const u32x4*)&Qs[arow * QP + akb + 16];
                const int brow = nt * 16 + laneM;
                const int bkb  = kc * 32 + laneHi * 8;
                Frag16 fb;
                fb.q[0] = *(const u32x4*)&Ks[brow * QP + bkb];
                fb.q[1] = *(const u32x4*)&Ks[brow * QP + bkb + 16];
                acc = __builtin_amdgcn_wmma_f32_16x16x32_bf16(
                          false, fa.v, false, fb.v, (short)0, acc, false, false);
            }
            const int rowbase = mt * 16 + laneHi * 8;
            const int col     = nt * 16 + laneM;
            #pragma unroll
            for (int j = 0; j < 4; ++j) {
                const int r0 = rowbase + 2 * j;
                float m0 = (r0     >= col) ? acc[2 * j]     : 0.0f;
                float m1 = (r0 + 1 >= col) ? acc[2 * j + 1] : 0.0f;
                unsigned int p = pk2bf(m0, m1);
                Ss[r0 * SP_ + col]       = (unsigned short)(p & 0xFFFFu);
                Ss[(r0 + 1) * SP_ + col] = (unsigned short)(p >> 16);
            }
        }
        __syncthreads();

        // ---- phase 2: O = S @ V  +  Q @ H ; 8 tiles, 1/wave ----
        {
            const int mt = wave >> 1, nt = wave & 1;
            v8f acc = 0.0f;
            // intra-chunk: S (64x64) @ V (64x32)
            #pragma unroll
            for (int kc = 0; kc < 2; ++kc) {
                const int arow = mt * 16 + laneM;
                const int akb  = kc * 32 + laneHi * 8;
                Frag16 fa;
                fa.q[0] = *(const u32x4*)&Ss[arow * SP_ + akb];
                fa.q[1] = *(const u32x4*)&Ss[arow * SP_ + akb + 16];
                const int brow = nt * 16 + laneM;
                const int bkb  = kc * 32 + laneHi * 8;
                Frag16 fb;
                fb.q[0] = *(const u32x4*)&Vt[brow * VTP + bkb];
                fb.q[1] = *(const u32x4*)&Vt[brow * VTP + bkb + 16];
                acc = __builtin_amdgcn_wmma_f32_16x16x32_bf16(
                          false, fa.v, false, fb.v, (short)0, acc, false, false);
            }
            // inter-chunk: Q (64x128) @ H (128x32), B frags from bf16 shadow
            #pragma unroll
            for (int kc = 0; kc < 4; ++kc) {
                const int arow = mt * 16 + laneM;
                const int akb  = kc * 32 + laneHi * 8;
                Frag16 fa;
                fa.q[0] = *(const u32x4*)&Qs[arow * QP + akb];
                fa.q[1] = *(const u32x4*)&Qs[arow * QP + akb + 16];
                const int hrow = nt * 16 + laneM;
                const int hkb  = kc * 32 + laneHi * 8;
                Frag16 fb;
                fb.q[0] = *(const u32x4*)&Hb[hrow * HBP + hkb];
                fb.q[1] = *(const u32x4*)&Hb[hrow * HBP + hkb + 16];
                acc = __builtin_amdgcn_wmma_f32_16x16x32_bf16(
                          false, fa.v, false, fb.v, (short)0, acc, false, false);
            }
            // store O tile (f32 output)
            #pragma unroll
            for (int r = 0; r < 8; ++r) {
                const int row = s0 + mt * 16 + r + laneHi * 8;
                const int col = slice * DV + nt * 16 + laneM;
                out[baseBH + (size_t)row * rowStride + col] = acc[r];
            }
        }
        __syncthreads();   // all Q@H reads of Hb done before the update

        // ---- phase 3: H += K^T @ V ; 16 tiles (8 dk x 2 dv), 2/wave ----
        #pragma unroll
        for (int t = 0; t < 2; ++t) {
            const int tile = wave + t * 8;
            const int mtH = tile >> 1, ntH = tile & 1;
            const int dvcol  = ntH * 16 + laneM;
            const int dkbase = mtH * 16 + laneHi * 8;
            v8f acc;
            #pragma unroll
            for (int r = 0; r < 8; ++r) acc[r] = Ht[dvcol * HP + dkbase + r];
            #pragma unroll
            for (int kc = 0; kc < 2; ++kc) {
                const int arow = mtH * 16 + laneM;           // dk row of K^T
                const int akb  = kc * 32 + laneHi * 8;
                Frag16 fa;
                fa.q[0] = *(const u32x4*)&Kt[arow * KTP + akb];
                fa.q[1] = *(const u32x4*)&Kt[arow * KTP + akb + 16];
                const int brow = ntH * 16 + laneM;           // dv row of V^T
                const int bkb  = kc * 32 + laneHi * 8;
                Frag16 fb;
                fb.q[0] = *(const u32x4*)&Vt[brow * VTP + bkb];
                fb.q[1] = *(const u32x4*)&Vt[brow * VTP + bkb + 16];
                acc = __builtin_amdgcn_wmma_f32_16x16x32_bf16(
                          false, fa.v, false, fb.v, (short)0, acc, false, false);
            }
            // write back: f32 master + packed bf16 shadow
            #pragma unroll
            for (int r = 0; r < 8; ++r) Ht[dvcol * HP + dkbase + r] = acc[r];
            #pragma unroll
            for (int j = 0; j < 4; ++j) {
                unsigned int p = pk2bf(acc[2 * j], acc[2 * j + 1]);
                *(unsigned int*)&Hb[dvcol * HBP + dkbase + 2 * j] = p;
            }
        }
    }
}

extern "C" void kernel_launch(void* const* d_in, const int* in_sizes, int n_in,
                              void* d_out, int out_size, void* d_ws, size_t ws_size,
                              hipStream_t stream) {
    (void)in_sizes; (void)n_in; (void)out_size; (void)d_ws; (void)ws_size;
    const float* q = (const float*)d_in[0];
    const float* k = (const float*)d_in[1];
    const float* v = (const float*)d_in[2];
    float* out = (float*)d_out;
    dim3 grid(B_ * H_ * NSL);   // 256 workgroups: (b, h, dv-slice)
    dim3 block(256);            // 8 waves
    linattn_fused_chunk_kernel<<<grid, block, 0, stream>>>(q, k, v, out);
}
